// MHAttentionMap_343597384027
// MI455X (gfx1250) — matrix operand order repaired and checked by reading, not compiled, not run on hardware
//
#include <hip/hip_runtime.h>
#include <hip/hip_bf16.h>
#include <math.h>

// Problem dims
#define BB   16
#define QQ   100
#define QP   112          // Q padded to multiple of 16
#define DD   256
#define NH   8
#define DHD  32
#define HW   4096
#define COLS (NH*HW)      // 32768 joint-softmax width
#define NORM_F 0.1767766952966369f   // (256/8)^-0.5

typedef __attribute__((ext_vector_type(16))) __bf16       v16bf;
typedef __attribute__((ext_vector_type(8)))  float        v8f;
typedef __attribute__((ext_vector_type(4)))  unsigned int u32x4;

struct B32x2 { u32x4 lo, hi; };

static __device__ __forceinline__ v16bf make_ab(u32x4 lo, u32x4 hi) {
  B32x2 t{lo, hi};
  return __builtin_bit_cast(v16bf, t);
}

// fp32 -> bf16 round-to-nearest-even
static __device__ __forceinline__ unsigned short f2bf(float f) {
  unsigned int u = __float_as_uint(f);
  u += 0x7fffu + ((u >> 16) & 1u);
  return (unsigned short)(u >> 16);
}

// ---------------- K0a: elementwise fp32 -> bf16 ----------------
__global__ void cvt_f32_bf16(const float* __restrict__ src,
                             unsigned short* __restrict__ dst, int n) {
  int i = blockIdx.x * blockDim.x + threadIdx.x;
  if (i < n) dst[i] = f2bf(src[i]);
}

// ---------------- K0b: transpose k [b][c][hw] f32 -> kt [b][hw][c] bf16 ----
__global__ void transpose_k(const float* __restrict__ k,
                            unsigned short* __restrict__ kt) {
  __shared__ float tile[32][33];
  const int b  = blockIdx.z;
  const int c0 = blockIdx.y * 32;
  const int h0 = blockIdx.x * 32;
  const int tx = threadIdx.x, ty = threadIdx.y;       // 32 x 8
  const float* kb = k + (size_t)b * DD * HW;
  for (int j = 0; j < 32; j += 8)
    tile[ty + j][tx] = kb[(size_t)(c0 + ty + j) * HW + h0 + tx];
  __syncthreads();
  unsigned short* ktb = kt + (size_t)b * HW * DD;
  for (int j = 0; j < 32; j += 8)
    ktb[(size_t)(h0 + ty + j) * DD + c0 + tx] = f2bf(tile[tx][ty + j]);
}

// ---------------- K1: qp = NORM*(q @ Wq^T + bq), bf16 out, rows padded ----
// One wave per 16x16 tile; M = padded (b,q) rows (1792), N = 256, K = 256.
__global__ void __launch_bounds__(256) qproj_wmma(
    const unsigned short* __restrict__ qbf,   // [1600][256]
    const unsigned short* __restrict__ wqbf,  // [256][256] (row e, col c)
    const float* __restrict__ bq,
    unsigned short* __restrict__ qp) {        // [1792][256]
  const int lane = threadIdx.x & 31;
  const int wave = threadIdx.x >> 5;
  const int tile = blockIdx.x * 8 + wave;     // 1792 tiles
  const int mt = tile >> 4;                   // 112 m-tiles
  const int nt = tile & 15;                   // 16 n-tiles
  const int l15 = lane & 15;
  const int hi  = lane >> 4;

  // A row (padded row -> clamp source q index)
  const int rr = mt * 16 + l15;
  const int b  = rr / QP;
  int q = rr % QP; if (q > QQ - 1) q = QQ - 1;
  const unsigned short* arow = qbf + (size_t)(b * QQ + q) * DD;

  const int ncol = nt * 16 + l15;
  const unsigned short* brow = wqbf + (size_t)ncol * DD;

  const float bias = bq[ncol];
  v8f acc; for (int i = 0; i < 8; ++i) acc[i] = bias;

  for (int kk = 0; kk < DD; kk += 32) {
    const u32x4* ap = (const u32x4*)(arow + kk + hi * 8);   // K {0..7,16..23}/{8..15,24..31}
    const u32x4* bp = (const u32x4*)(brow + kk + hi * 8);   // K {0..7,16..23}/{8..15,24..31}
    v16bf a  = make_ab(ap[0], ap[2]);
    v16bf bm = make_ab(bp[0], bp[2]);
    acc = __builtin_amdgcn_wmma_f32_16x16x32_bf16(false, a, false, bm,
                                                  (short)0, acc, false, false);
  }
  for (int r = 0; r < 8; ++r) {
    const int row = mt * 16 + r + hi * 8;                   // C: m = r (+8 upper half)
    qp[(size_t)row * DD + ncol] = f2bf(acc[r] * NORM_F);
  }
}

// ---------------- K2: kp[b][n][hw][dh] = bf16(Wk @ k + bk) ----------------
// M = hw (4096), N = e (256), K = c (256), per batch.
__global__ void __launch_bounds__(256) kproj_wmma(
    const unsigned short* __restrict__ ktbf,  // [b][hw][c] bf16
    const unsigned short* __restrict__ wkbf,  // [e][c] bf16
    const float* __restrict__ bk,
    unsigned short* __restrict__ kp) {        // [b][n][hw][dh] bf16
  const int lane = threadIdx.x & 31;
  const int wave = threadIdx.x >> 5;
  int tile = blockIdx.x * 8 + wave;           // 65536 tiles
  const int nt = tile & 15;  tile >>= 4;      // e tile
  const int ht = tile & 255; tile >>= 8;      // hw tile
  const int b  = tile;
  const int l15 = lane & 15;
  const int hi  = lane >> 4;

  const unsigned short* arow = ktbf + ((size_t)b * HW + ht * 16 + l15) * DD;
  const int ecol = nt * 16 + l15;
  const unsigned short* brow = wkbf + (size_t)ecol * DD;

  const float bias = bk[ecol];
  v8f acc; for (int i = 0; i < 8; ++i) acc[i] = bias;

  for (int kk = 0; kk < DD; kk += 32) {
    __builtin_prefetch(arow + kk + 64, 0, 0);               // global_prefetch_b8
    const u32x4* ap = (const u32x4*)(arow + kk + hi * 8);
    const u32x4* bp = (const u32x4*)(brow + kk + hi * 8);
    v16bf a  = make_ab(ap[0], ap[2]);
    v16bf bm = make_ab(bp[0], bp[2]);
    acc = __builtin_amdgcn_wmma_f32_16x16x32_bf16(false, a, false, bm,
                                                  (short)0, acc, false, false);
  }
  const int head = ecol >> 5, dh = ecol & 31;
  for (int r = 0; r < 8; ++r) {
    const int hw = ht * 16 + r + hi * 8;
    kp[((size_t)(b * NH + head) * HW + hw) * DHD + dh] = f2bf(acc[r]);
  }
}

// ---------------- K3: logits tile = qp(16x32) x kp(32x16), mask, -> out ---
__global__ void __launch_bounds__(256) attn_logits_wmma(
    const unsigned short* __restrict__ qp,    // [1792][256] bf16 (already *NORM)
    const unsigned short* __restrict__ kp,    // [b][n][hw][dh] bf16
    const unsigned char* __restrict__ mask,   // [b][hw] bool bytes
    float* __restrict__ out) {                // [b][q][n][hw] f32
  const int lane = threadIdx.x & 31;
  const int wave = threadIdx.x >> 5;
  int tile = blockIdx.x * 8 + wave;           // 229376 tiles
  const int ht   = tile & 255; tile >>= 8;    // hw tile
  const int head = tile & 7;   tile >>= 3;
  const int qt   = tile % 7;
  const int b    = tile / 7;
  const int l15 = lane & 15;
  const int hi  = lane >> 4;

  const unsigned short* arow = qp + (size_t)(b * QP + qt * 16 + l15) * DD + head * DHD;
  const unsigned short* brow = kp + ((size_t)(b * NH + head) * HW + ht * 16 + l15) * DHD;
  const u32x4* ap = (const u32x4*)(arow + hi * 8);
  const u32x4* bp = (const u32x4*)(brow + hi * 8);
  v16bf a  = make_ab(ap[0], ap[2]);
  v16bf bm = make_ab(bp[0], bp[2]);
  v8f acc = {};
  acc = __builtin_amdgcn_wmma_f32_16x16x32_bf16(false, a, false, bm,
                                                (short)0, acc, false, false);

  const int hw = ht * 16 + l15;
  const bool msk = mask[(size_t)b * HW + hw] != 0;
  const float NEG = -__builtin_inff();
  for (int r = 0; r < 8; ++r) {
    const int q = qt * 16 + r + hi * 8;
    if (q < QQ)
      out[((size_t)(b * QQ + q) * NH + head) * HW + hw] = msk ? NEG : acc[r];
  }
}

// ---------------- K4: per-row online (max, sumexp) over 32768 cols --------
__global__ void __launch_bounds__(256) row_stats_kernel(
    const float* __restrict__ out, float* __restrict__ stats) {
  const int row = blockIdx.x;                 // 1600 rows
  const float* p = out + (size_t)row * COLS;
  const float NEG = -__builtin_inff();
  float m = NEG, s = 0.f;
  for (int i = threadIdx.x; i < COLS; i += 256) {
    const float x = p[i];
    if (x == NEG) continue;                   // masked
    if (x > m) { s = s * __expf(m - x) + 1.f; m = x; }
    else       { s += __expf(x - m); }
  }
  __shared__ float sm[256], ss[256];
  sm[threadIdx.x] = m; ss[threadIdx.x] = s;
  __syncthreads();
  for (int w = 128; w > 0; w >>= 1) {
    if (threadIdx.x < w) {
      const float m1 = sm[threadIdx.x],     s1 = ss[threadIdx.x];
      const float m2 = sm[threadIdx.x + w], s2 = ss[threadIdx.x + w];
      const float mm = fmaxf(m1, m2);
      if (mm != NEG) {
        float ns = 0.f;
        if (m1 != NEG) ns += s1 * __expf(m1 - mm);
        if (m2 != NEG) ns += s2 * __expf(m2 - mm);
        sm[threadIdx.x] = mm; ss[threadIdx.x] = ns;
      }
    }
    __syncthreads();
  }
  if (threadIdx.x == 0) {
    stats[row * 2]     = sm[0];
    stats[row * 2 + 1] = (ss[0] > 0.f) ? 1.f / ss[0] : 0.f;
  }
}

// ---------------- K5: out = exp(x - m) * inv_s, vectorized ----------------
__global__ void __launch_bounds__(256) softmax_norm_kernel(
    float* __restrict__ out, const float* __restrict__ stats, long long n4) {
  const long long i = (long long)blockIdx.x * blockDim.x + threadIdx.x;
  if (i >= n4) return;
  float4* o = (float4*)out;
  float4 v = o[i];
  const int row = (int)((i * 4) >> 15);       // COLS = 2^15, 4 | 32768
  const float m  = stats[row * 2];
  const float is = stats[row * 2 + 1];
  v.x = __expf(v.x - m) * is;
  v.y = __expf(v.y - m) * is;
  v.z = __expf(v.z - m) * is;
  v.w = __expf(v.w - m) * is;
  o[i] = v;
}

extern "C" void kernel_launch(void* const* d_in, const int* in_sizes, int n_in,
                              void* d_out, int out_size, void* d_ws, size_t ws_size,
                              hipStream_t stream) {
  const float*         q    = (const float*)d_in[0];         // [16,100,256]
  const float*         k    = (const float*)d_in[1];         // [16,256,64,64]
  const unsigned char* mask = (const unsigned char*)d_in[2]; // [16,64,64] bool
  const float*         Wq   = (const float*)d_in[3];         // [256,256]
  const float*         bq   = (const float*)d_in[4];         // [256]
  const float*         Wk   = (const float*)d_in[5];         // [256,256]
  const float*         bk   = (const float*)d_in[6];         // [256]
  float* out = (float*)d_out;                                // [16,100,8,64,64]

  // Workspace layout (bytes), all regions 256B aligned
  char* ws = (char*)d_ws;
  unsigned short* qbf  = (unsigned short*)(ws + 0);          // 1600*256 bf16
  unsigned short* wqbf = (unsigned short*)(ws + 819200);     // 256*256 bf16
  unsigned short* wkbf = (unsigned short*)(ws + 950272);     // 256*256 bf16
  unsigned short* qp   = (unsigned short*)(ws + 1081344);    // 1792*256 bf16
  unsigned short* ktbf = (unsigned short*)(ws + 1998848);    // 16*4096*256 bf16
  unsigned short* kpbf = (unsigned short*)(ws + 35553280);   // 16*8*4096*32 bf16
  float*          stats= (float*)(ws + 69107712);            // 1600*2 f32

  cvt_f32_bf16<<<(409600 + 255) / 256, 256, 0, stream>>>(q, qbf, 409600);
  cvt_f32_bf16<<<(65536 + 255) / 256, 256, 0, stream>>>(Wq, wqbf, 65536);
  cvt_f32_bf16<<<(65536 + 255) / 256, 256, 0, stream>>>(Wk, wkbf, 65536);
  transpose_k<<<dim3(HW / 32, DD / 32, BB), dim3(32, 8), 0, stream>>>(k, ktbf);

  qproj_wmma<<<224, 256, 0, stream>>>(qbf, wqbf, bq, qp);        // 1792 tiles
  kproj_wmma<<<8192, 256, 0, stream>>>(ktbf, wkbf, bk, kpbf);    // 65536 tiles
  attn_logits_wmma<<<28672, 256, 0, stream>>>(qp, kpbf, mask, out); // 229376 tiles
  row_stats_kernel<<<1600, 256, 0, stream>>>(out, stats);
  softmax_norm_kernel<<<51200, 256, 0, stream>>>(out, stats, 13107200LL);
}
